// GroupLinear_38259568673602
// MI455X (gfx1250) — matrix operand order repaired and checked by reading, not compiled, not run on hardware
//
#include <hip/hip_runtime.h>
#include <hip/hip_bf16.h>
#include <stddef.h>

// Problem constants (from the reference)
#define BB    8
#define SS    2048
#define G     8
#define DIN   1024
#define DOUT  1024
#define NTOK  (BB * SS)            // 16384 tokens
#define NPERM (NTOK + G * 16)      // per-group padding to multiples of 16

typedef __bf16 bf16x16 __attribute__((ext_vector_type(16)));
typedef __bf16 bf16x8  __attribute__((ext_vector_type(8)));
typedef float  f32x8   __attribute__((ext_vector_type(8)));

// ---------------------------------------------------------------------------
// Workspace layout (ints): [0..7] counts, [8..15] cursors, [16..24] padoff,
// [32..] perm (NPERM ints). bf16 W copy at byte offset 128KB when ws allows.
// ---------------------------------------------------------------------------

__global__ void gl_init_kernel(int* __restrict__ meta, int* __restrict__ perm, int nperm) {
    int i = blockIdx.x * blockDim.x + threadIdx.x;
    if (i < 32) meta[i] = 0;
    if (i < nperm) perm[i] = -1;
}

__global__ void gl_hist_kernel(const int* __restrict__ gb, int* __restrict__ counts, int n) {
    int i = blockIdx.x * blockDim.x + threadIdx.x;
    if (i < n) atomicAdd(&counts[gb[i]], 1);
}

__global__ void gl_scan_kernel(const int* __restrict__ counts,
                               int* __restrict__ padoff, int* __restrict__ cursor) {
    if (threadIdx.x == 0) {
        int run = 0;
        for (int g = 0; g < G; ++g) {
            padoff[g] = run;
            run += ((counts[g] + 15) >> 4) << 4;   // pad each segment to 16
            cursor[g] = 0;
        }
        padoff[G] = run;
    }
}

__global__ void gl_scatter_kernel(const int* __restrict__ gb, int* __restrict__ cursor,
                                  const int* __restrict__ padoff, int* __restrict__ perm, int n) {
    int i = blockIdx.x * blockDim.x + threadIdx.x;
    if (i < n) {
        int g = gb[i];
        int pos = atomicAdd(&cursor[g], 1);
        perm[padoff[g] + pos] = i;
    }
}

// One-time f32 -> bf16 conversion of W (W then lives in L2: 16 MB << 192 MB).
__global__ void gl_wcvt_kernel(const float* __restrict__ Wf, __bf16* __restrict__ Wbf, int n4) {
    int i = blockIdx.x * blockDim.x + threadIdx.x;
    if (i < n4) {
        float4 v = ((const float4*)Wf)[i];
        __bf16* p = Wbf + (size_t)i * 4;
        p[0] = (__bf16)v.x; p[1] = (__bf16)v.y; p[2] = (__bf16)v.z; p[3] = (__bf16)v.w;
    }
}

// ---------------------------------------------------------------------------
// WMMA GEMM: block = 256 threads = 8 wave32. Block handles one group-uniform
// 16-token tile x 512 outputs; each wave = 16x64 (4 accumulators), K-loop 32.
// ---------------------------------------------------------------------------
template <bool USE_WBF>
__global__ __launch_bounds__(256)
void gl_gemm_kernel(const float* __restrict__ x,
                    const int*   __restrict__ gb,
                    const float* __restrict__ Wf,
                    const __bf16* __restrict__ Wbf,
                    const float* __restrict__ bias,
                    const int*   __restrict__ perm,
                    const int*   __restrict__ padoff,
                    float*       __restrict__ out) {
    __shared__ __bf16 xs[16 * DIN];   // 32 KB bf16 x-tile
    __shared__ int sm_tok[16];
    __shared__ int sm_g;

    const int tile   = blockIdx.x;
    const int ntiles = padoff[G] >> 4;
    if (tile >= ntiles) return;

    const int tid = threadIdx.x;
    if (tid < 16) {
        int t = perm[tile * 16 + tid];
        sm_tok[tid] = t;
        if (tid == 0) sm_g = gb[t];   // first row of a tile is always a real token
    }
    __syncthreads();
    const int g = sm_g;

    // Gather + convert x tile into LDS (16 float4 iterations per thread).
    for (int i = tid; i < 16 * (DIN / 4); i += 256) {
        int m  = i >> 8;              // 256 float4 per row
        int c4 = i & 255;
        int t  = sm_tok[m];
        const float4 v = ((const float4*)(x + (size_t)(t < 0 ? 0 : t) * DIN))[c4];
        __bf16* p = &xs[m * DIN + c4 * 4];
        p[0] = (__bf16)v.x; p[1] = (__bf16)v.y; p[2] = (__bf16)v.z; p[3] = (__bf16)v.w;
    }
    __syncthreads();

    const int wave = tid >> 5;
    const int lane = tid & 31;
    const int half = lane >> 4;       // K-half selector per ISA 16-bit layouts
    const int nloc = lane & 15;
    const int n0   = blockIdx.y * 512 + wave * 64;

    f32x8 acc0 = {}, acc1 = {}, acc2 = {}, acc3 = {};

    const __bf16* wb_base = Wbf + (size_t)g * DOUT * DIN;
    const float*  wf_base = Wf  + (size_t)g * DOUT * DIN;

    for (int k0 = 0; k0 < DIN; k0 += 32) {
        // A fragment (16x32 bf16): lane half h holds contiguous K run [16h,16h+16)
        const __bf16* ap = &xs[nloc * DIN + k0];
        bf16x8 alo = *(const bf16x8*)(ap + half * 16);
        bf16x8 ahi = *(const bf16x8*)(ap + half * 16 + 8);
        bf16x16 a = __builtin_shufflevector(alo, ahi,
                                            0, 1, 2, 3, 4, 5, 6, 7,
                                            8, 9, 10, 11, 12, 13, 14, 15);

        // B fragments (32x16 bf16): lane half h holds contiguous K run [16h,16h+16)
        const int koff = k0 + half * 16;
        bf16x16 b0, b1, b2, b3;
        if (USE_WBF) {
            const __bf16* wp = wb_base + (size_t)(n0 + nloc) * DIN + koff;
            b0 = *(const bf16x16*)(wp + 0 * 16 * DIN);
            b1 = *(const bf16x16*)(wp + 1 * 16 * DIN);
            b2 = *(const bf16x16*)(wp + 2 * 16 * DIN);
            b3 = *(const bf16x16*)(wp + 3 * 16 * DIN);
            if (k0 + 64 < DIN) __builtin_prefetch(wp + 64, 0, 0);  // global_prefetch next chunk
        } else {
            const float* wp = wf_base + (size_t)(n0 + nloc) * DIN + koff;
#pragma unroll
            for (int i = 0; i < 16; ++i) {
                b0[i] = (__bf16)wp[0 * 16 * DIN + i];
                b1[i] = (__bf16)wp[1 * 16 * DIN + i];
                b2[i] = (__bf16)wp[2 * 16 * DIN + i];
                b3[i] = (__bf16)wp[3 * 16 * DIN + i];
            }
            if (k0 + 64 < DIN) __builtin_prefetch(wp + 64, 0, 0);
        }

        // D = A*B + C  (v_wmma_f32_16x16x32_bf16)
        acc0 = __builtin_amdgcn_wmma_f32_16x16x32_bf16(false, a, false, b0, (short)0, acc0, false, false);
        acc1 = __builtin_amdgcn_wmma_f32_16x16x32_bf16(false, a, false, b1, (short)0, acc1, false, false);
        acc2 = __builtin_amdgcn_wmma_f32_16x16x32_bf16(false, a, false, b2, (short)0, acc2, false, false);
        acc3 = __builtin_amdgcn_wmma_f32_16x16x32_bf16(false, a, false, b3, (short)0, acc3, false, false);
    }

    // Epilogue: C/D layout is VGPR r -> (M = r + 8*half, N = lane&15). Add bias, scatter.
    const float bi0 = bias[g * DOUT + n0 + 0  + nloc];
    const float bi1 = bias[g * DOUT + n0 + 16 + nloc];
    const float bi2 = bias[g * DOUT + n0 + 32 + nloc];
    const float bi3 = bias[g * DOUT + n0 + 48 + nloc];
#pragma unroll
    for (int r = 0; r < 8; ++r) {
        int m = r + half * 8;
        int t = sm_tok[m];
        if (t >= 0) {
            float* o = out + (size_t)t * DOUT + n0 + nloc;
            o[0]  = acc0[r] + bi0;
            o[16] = acc1[r] + bi1;
            o[32] = acc2[r] + bi2;
            o[48] = acc3[r] + bi3;
        }
    }
}

// ---------------------------------------------------------------------------
extern "C" void kernel_launch(void* const* d_in, const int* in_sizes, int n_in,
                              void* d_out, int out_size, void* d_ws, size_t ws_size,
                              hipStream_t stream) {
    const float* x    = (const float*)d_in[0];   // [B,S,DIN] f32
    const int*   gb   = (const int*)d_in[1];     // [B,S] int32
    const float* W    = (const float*)d_in[2];   // [G, DIN*DOUT] f32 (row g -> (DOUT,DIN))
    const float* bias = (const float*)d_in[3];   // [G, DOUT] f32
    float* out = (float*)d_out;

    int* meta   = (int*)d_ws;
    int* counts = meta;
    int* cursor = meta + 8;
    int* padoff = meta + 16;
    int* perm   = meta + 32;

    const size_t WBF_OFF   = 128 * 1024;
    const size_t WBF_BYTES = (size_t)G * DIN * DOUT * 2;
    const bool use_wbf = ws_size >= WBF_OFF + WBF_BYTES;
    __bf16* Wbf = (__bf16*)((char*)d_ws + WBF_OFF);

    gl_init_kernel<<<(NPERM + 255) / 256, 256, 0, stream>>>(meta, perm, NPERM);
    gl_hist_kernel<<<(NTOK + 255) / 256, 256, 0, stream>>>(gb, counts, NTOK);
    gl_scan_kernel<<<1, 32, 0, stream>>>(counts, padoff, cursor);
    gl_scatter_kernel<<<(NTOK + 255) / 256, 256, 0, stream>>>(gb, cursor, padoff, perm, NTOK);
    if (use_wbf) {
        const int n4 = G * DIN * DOUT / 4;
        gl_wcvt_kernel<<<(n4 + 255) / 256, 256, 0, stream>>>(W, Wbf, n4);
    }

    // Worst-case padded tile count: 1024 full tiles + up to 1 partial per group.
    dim3 grid((unsigned)(NTOK / 16 + G), (unsigned)(DOUT / 512));
    if (use_wbf)
        gl_gemm_kernel<true><<<grid, 256, 0, stream>>>(x, gb, W, Wbf, bias, perm, padoff, out);
    else
        gl_gemm_kernel<false><<<grid, 256, 0, stream>>>(x, gb, W, Wbf, bias, perm, padoff, out);
}
